// WindowAttention_17575006175665
// MI455X (gfx1250) — compile-verified, matrix operand order checked by reading, not run on hardware
//
#include <hip/hip_runtime.h>
#include <hip/hip_bf16.h>
#include <hip/hip_fp16.h>

// ---------------------------------------------------------------------------
// Window attention (Swin) for MI455X / gfx1250.
//   B=4096 windows, N=49 tokens (pad->64), C=256, H=8 heads, hd=32.
//   All GEMMs run on v_wmma_f32_16x16x32_f16. One block per window, 8 waves.
//   Weight GEMMs are nj-outer so each weight fragment is read from L2 once
//   per block; A-fragments are re-read from LDS (cheap, 64 banks/WGP).
// ---------------------------------------------------------------------------

typedef __attribute__((ext_vector_type(16))) _Float16 v16h;
typedef __attribute__((ext_vector_type(8)))  float    v8f;

#define DIM      256
#define NHEADS   8
#define HD       32
#define NTOK     49
#define NPAD     64
#define TABLE_N  169

// LDS element offsets (f16 elements)
//   xs / AO : [64][256]           at 0        (32 KB)
//   qkvs    : [3][64][256]        at 16384    (96 KB)  (q block, k block, v block)
//   Ps      : [8][64][64] overlay at 16384    (64 KB, aliases q+k blocks after barrier)
#define XS_OFF   0
#define QKV_OFF  (NPAD * DIM)            // 16384
#define LDS_HALF_ELEMS (NPAD * DIM + 3 * NPAD * DIM)   // 131072 bytes total

// ---------------------------------------------------------------------------
// WMMA fragment loaders (ISA 7.12.2 layouts, wave32).
// A-frag (16x32 f16): lane L: M = L%16; halves e: K = (e/8)*16 + (L/16)*8 + e%8
// B-frag (32x16 f16): lane L: N = L%16; same K pattern.
// C/D    (16x16 f32): vgpr v: M = v + 8*(L/16), N = L%16.
// ---------------------------------------------------------------------------

__device__ __forceinline__ v8f wmma_f16(v16h a, v16h b, v8f c) {
    return __builtin_amdgcn_wmma_f32_16x16x32_f16(
        /*neg_a=*/false, a, /*neg_b=*/false, b,
        /*c_mod=*/(short)0, c, /*reuse_a=*/false, /*reuse_b=*/false);
}

// Row-major [M][K] source (also serves B when the matrix is stored transposed,
// i.e. row-major [N][K]):  elem(m_or_n, k) = base[(m_or_n)*ld + k]
__device__ __forceinline__ v16h load_frag_mk(const _Float16* __restrict__ base, int ld) {
    const int lane = threadIdx.x & 31;
    const int m    = lane & 15;
    const int lh   = lane >> 4;
    const _Float16* p = base + m * ld + (lh << 3);
    v16h f;
#pragma unroll
    for (int e = 0; e < 16; ++e) {
        const int k = ((e >> 3) << 4) + (e & 7);   // + lh*8 folded into p
        f[e] = p[k];
    }
    return f;
}

// Row-major [K][N] source: B(k,n) = base[k*ld + n]
__device__ __forceinline__ v16h load_frag_kn(const _Float16* __restrict__ base, int ld) {
    const int lane = threadIdx.x & 31;
    const int n    = lane & 15;
    const int lh   = lane >> 4;
    v16h f;
#pragma unroll
    for (int e = 0; e < 16; ++e) {
        const int k = ((e >> 3) << 4) + (lh << 3) + (e & 7);
        f[e] = base[k * ld + n];
    }
    return f;
}

// ---------------------------------------------------------------------------
// Prep kernel: weights -> f16 (stay L2-resident, 192MB L2), bias gather.
// ---------------------------------------------------------------------------
__global__ __launch_bounds__(256) void wa_prep_kernel(
    const float* __restrict__ qkv_w, const float* __restrict__ proj_w,
    const float* __restrict__ bias_table, const int* __restrict__ rel_idx,
    _Float16* __restrict__ qkv_w16, _Float16* __restrict__ proj_w16,
    float* __restrict__ bias_ws) {
    const int i = blockIdx.x * blockDim.x + threadIdx.x;
    if (i < 3 * DIM * DIM)  qkv_w16[i]  = (_Float16)qkv_w[i];
    if (i < DIM * DIM)      proj_w16[i] = (_Float16)proj_w[i];
    if (i < NHEADS * NTOK * NTOK) {
        const int h  = i / (NTOK * NTOK);
        const int rm = i % (NTOK * NTOK);
        bias_ws[i] = bias_table[rel_idx[rm] * NHEADS + h];
    }
}

// ---------------------------------------------------------------------------
// Main kernel: one block per window.
// ---------------------------------------------------------------------------
__global__ __launch_bounds__(256, 2) void wa_attn_kernel(
    const float* __restrict__ x,
    const float* __restrict__ qkv_b, const float* __restrict__ proj_b,
    const _Float16* __restrict__ qkv_w16, const _Float16* __restrict__ proj_w16,
    const float* __restrict__ bias_ws,
    float* __restrict__ out) {

    extern __shared__ _Float16 smem[];
    _Float16* xs   = smem + XS_OFF;    // [64][256] x tile, later AO tile
    _Float16* qkvs = smem + QKV_OFF;   // [3][64][256]
    _Float16* Ps   = smem + QKV_OFF;   // [8][64][64] overlay on q+k blocks

    const int b    = blockIdx.x;
    const int tid  = threadIdx.x;
    const int wave = tid >> 5;
    const int lane = tid & 31;
    const int cl   = lane & 15;
    const int lh   = lane >> 4;

    // ---- Stage 0: load x[b] (49x256 f32) -> f16 LDS, zero-pad rows 49..63 --
    {
        const size_t xbase = (size_t)b * NTOK * DIM;
        for (int i = tid; i < NPAD * DIM; i += 256) {
            const int r = i >> 8, c = i & 255;
            const float v = (r < NTOK) ? x[xbase + (size_t)r * DIM + c] : 0.0f;
            xs[i] = (_Float16)v;
        }
    }
    __syncthreads();

    // ---- Stage 1: QKV GEMM  [64x256] x [256x768] -> qkvs [3][64][256] ------
    // 48 N-tiles, wave w owns N-tiles [6w, 6w+6).  nj-outer: each weight
    // fragment is pulled from L2 exactly once per block.
    for (int nj = 0; nj < 6; ++nj) {
        const int nt = wave * 6 + nj;
        v16h bfrag[8];
#pragma unroll
        for (int k8 = 0; k8 < 8; ++k8)
            bfrag[k8] = load_frag_mk(qkv_w16 + (size_t)nt * 16 * DIM + k8 * 32, DIM);
        const int col  = nt * 16 + cl;          // 0..767
        const int part = col >> 8;              // 0=q 1=k 2=v
        const int cc   = col & 255;
        const float bb = qkv_b[col];
        for (int mi = 0; mi < 4; ++mi) {
            v8f acc = {0.f, 0.f, 0.f, 0.f, 0.f, 0.f, 0.f, 0.f};
#pragma unroll
            for (int k8 = 0; k8 < 8; ++k8) {
                const v16h afrag = load_frag_mk(xs + mi * 16 * DIM + k8 * 32, DIM);
                acc = wmma_f16(afrag, bfrag[k8], acc);
            }
#pragma unroll
            for (int v = 0; v < 8; ++v) {
                const int row = mi * 16 + v + 8 * lh;
                qkvs[part * (NPAD * DIM) + row * DIM + cc] = (_Float16)(acc[v] + bb);
            }
        }
    }
    __syncthreads();

    // ---- Stage 2: per-head attention, wave == head ------------------------
    const int h = wave;
    const _Float16* qh = qkvs + 0 * (NPAD * DIM) + h * HD;   // ld = 256
    const _Float16* kh = qkvs + 1 * (NPAD * DIM) + h * HD;
    const _Float16* vh = qkvs + 2 * (NPAD * DIM) + h * HD;

    v8f S[4][4];
    {
        v16h kb[4];
#pragma unroll
        for (int ni = 0; ni < 4; ++ni)
            kb[ni] = load_frag_mk(kh + ni * 16 * DIM, DIM);   // B(k,n)=K[n][k]
#pragma unroll
        for (int mi = 0; mi < 4; ++mi) {
            const v16h qa = load_frag_mk(qh + mi * 16 * DIM, DIM);
            v8f z = {0.f, 0.f, 0.f, 0.f, 0.f, 0.f, 0.f, 0.f};
#pragma unroll
            for (int ni = 0; ni < 4; ++ni)
                S[mi][ni] = wmma_f16(qa, kb[ni], z);
        }
    }
    // All waves finished reading q/k blocks -> safe to overlay Ps on them.
    __syncthreads();

    // scale + bias + mask + row softmax (rows live on 16-lane halves)
    {
        const float scale = 0.17677669529663688f;   // 1/sqrt(32)
        const float* bh = bias_ws + h * (NTOK * NTOK);
#pragma unroll
        for (int mi = 0; mi < 4; ++mi) {
#pragma unroll
            for (int v = 0; v < 8; ++v) {
                const int r = mi * 16 + v + 8 * lh;
                float vals[4];
#pragma unroll
                for (int ni = 0; ni < 4; ++ni) {
                    const int c = ni * 16 + cl;
                    float s = -1e30f;
                    if (c < NTOK && r < NTOK)
                        s = S[mi][ni][v] * scale + bh[r * NTOK + c];
                    vals[ni] = s;
                }
                float mx = fmaxf(fmaxf(vals[0], vals[1]), fmaxf(vals[2], vals[3]));
#pragma unroll
                for (int off = 1; off < 16; off <<= 1)
                    mx = fmaxf(mx, __shfl_xor(mx, off, 32));
                float sum = 0.f;
#pragma unroll
                for (int ni = 0; ni < 4; ++ni) {
                    vals[ni] = __expf(vals[ni] - mx);
                    sum += vals[ni];
                }
#pragma unroll
                for (int off = 1; off < 16; off <<= 1)
                    sum += __shfl_xor(sum, off, 32);
                const float inv = 1.0f / sum;
#pragma unroll
                for (int ni = 0; ni < 4; ++ni)
                    Ps[h * (NPAD * NPAD) + r * NPAD + ni * 16 + cl] =
                        (_Float16)(vals[ni] * inv);
            }
        }
    }

    // ---- Stage 3: O = P @ V  -> AO (overlaid on xs), cols h*32..h*32+31 ---
    for (int mi = 0; mi < 4; ++mi) {
        v16h pa[2];
#pragma unroll
        for (int ks = 0; ks < 2; ++ks)
            pa[ks] = load_frag_mk(Ps + h * (NPAD * NPAD) + mi * 16 * NPAD + ks * 32, NPAD);
#pragma unroll
        for (int nj = 0; nj < 2; ++nj) {
            v8f acc = {0.f, 0.f, 0.f, 0.f, 0.f, 0.f, 0.f, 0.f};
#pragma unroll
            for (int ks = 0; ks < 2; ++ks) {
                const v16h vb = load_frag_kn(vh + ks * 32 * DIM + nj * 16, DIM);
                acc = wmma_f16(pa[ks], vb, acc);
            }
#pragma unroll
            for (int v = 0; v < 8; ++v) {
                const int row = mi * 16 + v + 8 * lh;
                xs[row * DIM + h * HD + nj * 16 + cl] = (_Float16)acc[v];
            }
        }
    }
    __syncthreads();

    // ---- Stage 4: out = AO @ proj_w^T + proj_b ----------------------------
    // 16 N-tiles, wave w owns N-tiles {2w, 2w+1}.  nj-outer as in stage 1.
    for (int nj = 0; nj < 2; ++nj) {
        const int nt = wave * 2 + nj;
        v16h bfrag[8];
#pragma unroll
        for (int k8 = 0; k8 < 8; ++k8)
            bfrag[k8] = load_frag_mk(proj_w16 + (size_t)nt * 16 * DIM + k8 * 32, DIM);
        const int col  = nt * 16 + cl;
        const float pb = proj_b[col];
        const size_t obase = (size_t)b * NTOK * DIM;
        for (int mi = 0; mi < 4; ++mi) {
            v8f acc = {0.f, 0.f, 0.f, 0.f, 0.f, 0.f, 0.f, 0.f};
#pragma unroll
            for (int k8 = 0; k8 < 8; ++k8) {
                const v16h afrag = load_frag_mk(xs + mi * 16 * DIM + k8 * 32, DIM);
                acc = wmma_f16(afrag, bfrag[k8], acc);
            }
#pragma unroll
            for (int v = 0; v < 8; ++v) {
                const int row = mi * 16 + v + 8 * lh;
                if (row < NTOK)
                    out[obase + (size_t)row * DIM + col] = acc[v] + pb;
            }
        }
    }
}

// ---------------------------------------------------------------------------
// Launch
// ---------------------------------------------------------------------------
extern "C" void kernel_launch(void* const* d_in, const int* in_sizes, int n_in,
                              void* d_out, int out_size, void* d_ws, size_t ws_size,
                              hipStream_t stream) {
    const float* x          = (const float*)d_in[0];
    const float* qkv_w      = (const float*)d_in[1];
    const float* qkv_b      = (const float*)d_in[2];
    const float* proj_w     = (const float*)d_in[3];
    const float* proj_b     = (const float*)d_in[4];
    const float* bias_table = (const float*)d_in[5];
    const int*   rel_idx    = (const int*)d_in[6];
    float* out = (float*)d_out;

    // Workspace carve-up
    char* ws = (char*)d_ws;
    _Float16* qkv_w16  = (_Float16*)(ws);                 // 768*256*2 = 393216 B
    _Float16* proj_w16 = (_Float16*)(ws + 393216);        // 256*256*2 = 131072 B
    float*    bias_ws  = (float*)   (ws + 524288);        // 8*49*49*4 =  76832 B

    const int prep_elems = 3 * DIM * DIM;                 // 196608 (largest task)
    wa_prep_kernel<<<(prep_elems + 255) / 256, 256, 0, stream>>>(
        qkv_w, proj_w, bias_table, rel_idx, qkv_w16, proj_w16, bias_ws);

    const size_t lds_bytes = (size_t)LDS_HALF_ELEMS * sizeof(_Float16); // 128 KB
    wa_attn_kernel<<<4096, 256, lds_bytes, stream>>>(
        x, qkv_b, proj_b, qkv_w16, proj_w16, bias_ws, out);
}
